// KLayerGCNConv_62569083568438
// MI455X (gfx1250) — matrix operand order repaired and checked by reading, not compiled, not run on hardware
//
#include <hip/hip_runtime.h>
#include <hip/hip_bf16.h>

typedef __attribute__((ext_vector_type(16))) __bf16 v16bf;
typedef __attribute__((ext_vector_type(8)))  float  v8f;

#define IN_CH  128
#define OUT_CH 64
#define SELF_W 2.0f

// ---------------- kernel 1: degree init with self-loop weight ----------------
__global__ void k_deg_init(float* __restrict__ deg, int N) {
  int i = blockIdx.x * blockDim.x + threadIdx.x;
  if (i < N) deg[i] = SELF_W;
}

// ---------------- kernel 2: accumulate edge weights into degree --------------
__global__ void k_deg_edges(const int* __restrict__ ei, const float* __restrict__ ew,
                            float* __restrict__ deg, int E) {
  int e = blockIdx.x * blockDim.x + threadIdx.x;
  if (e < E) {
    int col = ei[(size_t)E + e];            // edge_index[1][e]
    atomicAdd(&deg[col], ew[e]);
  }
}

// ---------------- kernel 3: xw = x @ W via bf16 WMMA -------------------------
// One wave computes a 16x64 tile (K=128 = 4 chunks of 32, N=64 = 4 n-tiles).
// W is pre-swizzled into LDS in exact B-fragment order so each fragment is a
// single aligned 32-byte v16bf load (2x ds_load_b128) instead of 16 scalar
// ds_load_u16.  Results are staged through a per-wave LDS tile so global
// stores are coalesced global_store_b128 (consecutive lanes -> consecutive
// 128B) with a wave-uniform tail check instead of per-row EXEC predication.
__global__ void __launch_bounds__(256) k_gemm(const float* __restrict__ x,
                                              const float* __restrict__ W,
                                              float* __restrict__ xw,
                                              int N, int mtiles) {
  // B fragments: [kc(4)][nt(4)][lane(32)][e(16)] bf16, 16 KB total.
  // Element e of lane's fragment: K = kc*32 + (lane>>4)*8 + (e&7) + (e>>3)*16,
  //                               n = nt*16 + (lane&15).
  __shared__ __bf16 sB[4 * 4 * 32 * 16];
  __shared__ float  sT[8][16 * OUT_CH];     // per-wave 16x64 f32 staging, 32 KB

  for (int idx = threadIdx.x; idx < 4 * 4 * 32 * 16; idx += 256) {
    const int e  = idx & 15;
    const int ln = (idx >> 4) & 31;
    const int nt = (idx >> 9) & 3;
    const int kc = idx >> 11;
    const int k  = kc * 32 + (ln >> 4) * 8 + (e & 7) + (e >> 3) * 16;
    const int n  = nt * 16 + (ln & 15);
    sB[idx] = (__bf16)W[k * OUT_CH + n];
  }
  __syncthreads();                           // before any wave can exit

  const int wave = threadIdx.x >> 5;
  const int lane = threadIdx.x & 31;
  const int mtile = blockIdx.x * 8 + wave;   // wave-uniform
  if (mtile >= mtiles) return;               // whole-wave exit: EXEC stays all-1s

  const int mrow = lane & 15;
  const int hi   = lane >> 4;
  int grow = mtile * 16 + mrow;
  if (grow >= N) grow = N - 1;               // branchless clamp (tail only)
  const float* xr = x + (size_t)grow * IN_CH + hi * 8;

  v8f acc[4] = {};
  const __bf16* bbase = sB + (lane << 4);

#pragma unroll
  for (int kc = 0; kc < 4; ++kc) {
    const int k0 = kc * 32;
    // A fragment: two contiguous 8-float runs -> 4x global_load_b128 + cvt
    float af[16];
    *(float4*)(af + 0)  = *(const float4*)(xr + k0 + 0);
    *(float4*)(af + 4)  = *(const float4*)(xr + k0 + 4);
    *(float4*)(af + 8)  = *(const float4*)(xr + k0 + 16);
    *(float4*)(af + 12) = *(const float4*)(xr + k0 + 20);
    v16bf a;
#pragma unroll
    for (int e = 0; e < 16; ++e) a[e] = (__bf16)af[e];

#pragma unroll
    for (int nt = 0; nt < 4; ++nt) {
      const v16bf b = *(const v16bf*)(bbase + ((kc * 4 + nt) << 9));
      acc[nt] = __builtin_amdgcn_wmma_f32_16x16x32_bf16(
          /*neg_a=*/false, a, /*neg_b=*/false, b,
          /*c_mod=*/(short)0, acc[nt], /*reuse_a=*/false, /*reuse_b=*/false);
    }
  }

  // stage tile in per-wave LDS: acc[nt][r] -> row (hi*8+r), col (nt*16+mrow)
  float* st = &sT[wave][0];
#pragma unroll
  for (int nt = 0; nt < 4; ++nt)
#pragma unroll
    for (int r = 0; r < 8; ++r)
      st[(hi * 8 + r) * OUT_CH + nt * 16 + mrow] = acc[nt][r];

  // per-wave private staging: same-wave LDS ops are in-order (DScnt);
  // barrier only stops compiler reordering.
  __builtin_amdgcn_wave_barrier();

  // coalesced writeback: lane -> (row = lane>>1, colbase = (lane&1)*32),
  // 8x float4 per lane; lanes 2k,2k+1 cover one full 256B row.
  const int r2 = lane >> 1;
  const int cb = (lane & 1) * 32;
  const int gm = mtile * 16 + r2;
  const bool full = (mtile * 16 + 16 <= N);  // wave-uniform except tail tile
  if (full || gm < N) {
    float* dst = xw + (size_t)gm * OUT_CH + cb;
    const float* src = st + r2 * OUT_CH + cb;
#pragma unroll
    for (int q = 0; q < 8; ++q)
      *(float4*)(dst + q * 4) = *(const float4*)(src + q * 4);
  }
}

// ---------------- kernel 4: dinv = deg > 0 ? rsqrt(deg) : 0 ------------------
__global__ void k_dinv(const float* __restrict__ deg, float* __restrict__ dinv, int N) {
  int i = blockIdx.x * blockDim.x + threadIdx.x;
  if (i < N) {
    float d = deg[i];
    dinv[i] = (d > 0.0f) ? rsqrtf(d) : 0.0f;
  }
}

// ---------------- kernel 5: out = b + selfloop_norm * xw  (float4) -----------
__global__ void k_out_init(const float* __restrict__ xw, const float* __restrict__ dinv,
                           const float* __restrict__ b, float* __restrict__ out, int N) {
  int idx = blockIdx.x * blockDim.x + threadIdx.x;   // one float4 per thread
  if (idx < N * (OUT_CH / 4)) {
    const int i  = idx >> 4;                          // node
    const int c4 = (idx & 15) * 4;                    // channel base
    const float di = dinv[i];
    const float s  = SELF_W * di * di;
    const float4 v  = *(const float4*)(xw + (size_t)i * OUT_CH + c4);
    const float4 bb = *(const float4*)(b + c4);
    float4 o;
    o.x = bb.x + s * v.x;
    o.y = bb.y + s * v.y;
    o.z = bb.z + s * v.z;
    o.w = bb.w + s * v.w;
    *(float4*)(out + (size_t)i * OUT_CH + c4) = o;
  }
}

// ---------------- kernel 6: edge scatter (dominant, bandwidth-bound) ---------
// 16 edges per 256-thread block; 16 threads per edge, each handling 4 channels
// via a float4 gather + 4 global_atomic_add_f32.  Edge metadata (row/col/norm)
// computed once per edge by a leader thread, broadcast through LDS.
__global__ void __launch_bounds__(256) k_scatter(const int* __restrict__ ei,
                                                 const float* __restrict__ ew,
                                                 const float* __restrict__ dinv,
                                                 const float* __restrict__ xw,
                                                 float* __restrict__ out, int E) {
  __shared__ float s_norm[16];
  __shared__ int   s_row[16];
  __shared__ int   s_col[16];

  const int t = threadIdx.x;
  const size_t ebase = (size_t)blockIdx.x * 16;

  if (t < 16) {
    size_t e = ebase + t;
    if (e < (size_t)E) {
      int r = ei[e];
      int c = ei[(size_t)E + e];
      s_row[t]  = r;
      s_col[t]  = c;
      s_norm[t] = dinv[r] * ew[e] * dinv[c];
    } else {
      s_row[t] = 0; s_col[t] = 0; s_norm[t] = 0.0f;
    }
  }
  __syncthreads();

  const int sub = t >> 4;        // edge slot 0..15
  const int c4  = (t & 15) * 4;  // channel base
  if (ebase + (size_t)sub >= (size_t)E) return;

  const float nrm = s_norm[sub];
  const int   row = s_row[sub];
  const int   col = s_col[sub];

  const float4 v = *(const float4*)(xw + (size_t)row * OUT_CH + c4);
  float* o = out + (size_t)col * OUT_CH + c4;
  atomicAdd(o + 0, nrm * v.x);
  atomicAdd(o + 1, nrm * v.y);
  atomicAdd(o + 2, nrm * v.z);
  atomicAdd(o + 3, nrm * v.w);
}

extern "C" void kernel_launch(void* const* d_in, const int* in_sizes, int n_in,
                              void* d_out, int out_size, void* d_ws, size_t ws_size,
                              hipStream_t stream) {
  const float* x  = (const float*)d_in[0];
  const int*   ei = (const int*)d_in[1];     // edge_index [2,E] (row-major)
  const float* ew = (const float*)d_in[2];
  const float* W  = (const float*)d_in[3];
  const float* b  = (const float*)d_in[4];
  float* out = (float*)d_out;

  const int N = in_sizes[0] / IN_CH;         // 50000
  const int E = in_sizes[1] / 2;             // 800000

  // workspace layout: deg[N] | dinv[N] | xw[N*64]
  char* ws = (char*)d_ws;
  float* deg = (float*)ws;
  size_t off = ((size_t)N * sizeof(float) + 255) & ~(size_t)255;
  float* dinv = (float*)(ws + off);
  off = (off + (size_t)N * sizeof(float) + 255) & ~(size_t)255;
  float* xw = (float*)(ws + off);

  k_deg_init<<<(N + 255) / 256, 256, 0, stream>>>(deg, N);
  k_deg_edges<<<(E + 255) / 256, 256, 0, stream>>>(ei, ew, deg, E);

  const int mtiles = (N + 15) / 16;
  k_gemm<<<(mtiles + 7) / 8, 256, 0, stream>>>(x, W, xw, N, mtiles);

  k_dinv<<<(N + 255) / 256, 256, 0, stream>>>(deg, dinv, N);
  k_out_init<<<((size_t)N * (OUT_CH / 4) + 255) / 256, 256, 0, stream>>>(xw, dinv, b, out, N);
  k_scatter<<<(E + 15) / 16, 256, 0, stream>>>(ei, ew, dinv, xw, out, E);
}
